// KMGCN_63634235457560
// MI455X (gfx1250) — matrix operand order repaired and checked by reading, not compiled, not run on hardware
//
#include <hip/hip_runtime.h>
#include <hip/hip_bf16.h>

typedef __attribute__((ext_vector_type(16))) _Float16 v16h;
typedef __attribute__((ext_vector_type(8)))  _Float16 v8h;
typedef __attribute__((ext_vector_type(8)))  float    v8f;

// Problem constants (match reference)
#define IN_DIM   128
#define HID      256
#define HID2     128
#define OUT_DIM  4
#define N_GRAPHS 64

// ---------------------------------------------------------------------------
// Degree + dinv
// ---------------------------------------------------------------------------
__global__ __launch_bounds__(256) void k_degree(const int* __restrict__ dst,
                                                float* __restrict__ deg, int nEdges) {
    int e = blockIdx.x * blockDim.x + threadIdx.x;
    if (e < nEdges) atomicAdd(&deg[dst[e]], 1.0f);
}

__global__ __launch_bounds__(256) void k_dinv(const float* __restrict__ deg,
                                              float* __restrict__ dinv, int nNodes) {
    int i = blockIdx.x * blockDim.x + threadIdx.x;
    if (i < nNodes) dinv[i] = rsqrtf(deg[i] + 1.0f);   // +1 self loop, deg >= 1
}

// ---------------------------------------------------------------------------
// f32 -> f16 conversions
// ---------------------------------------------------------------------------
__global__ __launch_bounds__(256) void k_cvt_f16(const float* __restrict__ in,
                                                 _Float16* __restrict__ out, int n) {
    int i = blockIdx.x * blockDim.x + threadIdx.x;
    if (i < n) out[i] = (_Float16)in[i];
}

// W[K][N] row-major f32 -> Wt[N][K] f16 (so each WMMA B-lane reads contiguous K)
template<int K, int N>
__global__ __launch_bounds__(256) void k_cvt_wt(const float* __restrict__ W,
                                                _Float16* __restrict__ Wt) {
    int i = blockIdx.x * blockDim.x + threadIdx.x;
    if (i < K * N) {
        int k = i / N, n = i % N;
        Wt[(size_t)n * K + k] = (_Float16)W[i];
    }
}

// ---------------------------------------------------------------------------
// WMMA GEMM: C[M,N] = A[M,K] * B[K,N]
//   A : f16 row-major [M,K]
//   Bt: f16 [N,K]  (row n holds column n of B)
//   C : f32 row-major [M,N]
// One wave32 per 16-row tile; A chunks preloaded and reused across all N tiles.
// ---------------------------------------------------------------------------
template<int K, int N>
__global__ __launch_bounds__(256) void k_wmma_gemm(const _Float16* __restrict__ A,
                                                   const _Float16* __restrict__ Bt,
                                                   float* __restrict__ C, int mtiles) {
    constexpr int KC = K / 32;   // k-chunks of 32
    constexpr int NT = N / 16;   // 16-wide n tiles
    const int lane = threadIdx.x & 31;
    const int wave = threadIdx.x >> 5;
    const int gw   = blockIdx.x * (blockDim.x >> 5) + wave;   // wave-uniform
    if (gw >= mtiles) return;                                  // whole wave exits: EXEC stays all-1

    const int m0 = gw << 4;
    const int r  = lane & 15;
    const int hi = lane >> 4;

    // A: lane r / r+16 reads row m0+r; half-group selects k sub-ranges
    v16h a[KC];
    const _Float16* arow = A + (size_t)(m0 + r) * K + hi * 8;
    #pragma unroll
    for (int kc = 0; kc < KC; ++kc) {
        v16h t;
        ((v8h*)&t)[0] = *(const v8h*)(arow + kc * 32);        // k = kc*32 + hi*8 + 0..7
        ((v8h*)&t)[1] = *(const v8h*)(arow + kc * 32 + 16);   // k = kc*32 + 16 + hi*8 + 0..7
        a[kc] = t;
    }

    #pragma unroll
    for (int nt = 0; nt < NT; ++nt) {
        v8f c = {};
        const _Float16* brow = Bt + (size_t)(nt * 16 + r) * K + hi * 8;
        #pragma unroll
        for (int kc = 0; kc < KC; ++kc) {
            v16h b;
            ((v8h*)&b)[0] = *(const v8h*)(brow + kc * 32);     // k = kc*32 + hi*8 + 0..7
            ((v8h*)&b)[1] = *(const v8h*)(brow + kc * 32 + 16); // k = kc*32 + 16 + hi*8 + 0..7
            c = __builtin_amdgcn_wmma_f32_16x16x32_f16(
                    false, a[kc], false, b, (short)0, c, false, false);
        }
        // C/D layout: VGPR v -> row m0 + v + 8*hi, col nt*16 + r
        #pragma unroll
        for (int v = 0; v < 8; ++v)
            C[(size_t)(m0 + v + hi * 8) * N + nt * 16 + r] = c[v];
    }
}

// ---------------------------------------------------------------------------
// Edge scatter: agg[dst] += h[src] * dinv[src]*dinv[dst]   (float4 per thread)
// ---------------------------------------------------------------------------
template<int F>
__global__ __launch_bounds__(256) void k_scatter(const float* __restrict__ h,
                                                 const int* __restrict__ src,
                                                 const int* __restrict__ dst,
                                                 const float* __restrict__ dinv,
                                                 float* __restrict__ agg, int nEdges) {
    constexpr int CH = F / 4;
    int tid = blockIdx.x * blockDim.x + threadIdx.x;
    if (tid >= nEdges * CH) return;
    int e = tid / CH;
    int c = tid - e * CH;
    int s = src[e], d = dst[e];
    if (c == 0 && e + 8 < nEdges) {                 // hide gather latency
        __builtin_prefetch(h + (size_t)src[e + 8] * F, 0, 0);
    }
    float nrm = dinv[s] * dinv[d];
    const float4 v = ((const float4*)(h + (size_t)s * F))[c];
    float* out = agg + (size_t)d * F + c * 4;
    atomicAdd(out + 0, v.x * nrm);
    atomicAdd(out + 1, v.y * nrm);
    atomicAdd(out + 2, v.z * nrm);
    atomicAdd(out + 3, v.w * nrm);
}

// ---------------------------------------------------------------------------
// Finalize layer 1: relu(agg + h*dinv^2 + b) -> f16 for next GEMM
// ---------------------------------------------------------------------------
template<int F>
__global__ __launch_bounds__(256) void k_finalize_f16(const float* __restrict__ agg,
                                                      const float* __restrict__ hlin,
                                                      const float* __restrict__ dinv,
                                                      const float* __restrict__ bias,
                                                      _Float16* __restrict__ out16,
                                                      int nNodes) {
    int i = blockIdx.x * blockDim.x + threadIdx.x;
    if (i >= nNodes * F) return;
    int node = i / F, f = i - node * F;
    float di = dinv[node];
    float v = agg[i] + hlin[i] * di * di + bias[f];
    out16[i] = (_Float16)fmaxf(v, 0.0f);
}

// ---------------------------------------------------------------------------
// Finalize layer 2 + mean-pool accumulation (atomics into [64, HID2])
// ---------------------------------------------------------------------------
__global__ __launch_bounds__(256) void k_finalize_pool(const float* __restrict__ agg,
                                                       const float* __restrict__ hlin,
                                                       const float* __restrict__ dinv,
                                                       const float* __restrict__ bias,
                                                       const int* __restrict__ batch,
                                                       float* __restrict__ pool,
                                                       int nNodes) {
    int i = blockIdx.x * blockDim.x + threadIdx.x;
    if (i >= nNodes * HID2) return;
    int node = i >> 7, f = i & (HID2 - 1);
    float di = dinv[node];
    float v = fmaxf(agg[i] + hlin[i] * di * di + bias[f], 0.0f);
    atomicAdd(&pool[(size_t)batch[node] * HID2 + f], v);
}

__global__ __launch_bounds__(256) void k_count(const int* __restrict__ batch,
                                               float* __restrict__ cnt, int nNodes) {
    int i = blockIdx.x * blockDim.x + threadIdx.x;
    if (i < nNodes) atomicAdd(&cnt[batch[i]], 1.0f);
}

// ---------------------------------------------------------------------------
// Final FC: out[g,o] = mean(pool[g]) @ Wfc + bfc   (64*4 = 256 threads, 1 block)
// ---------------------------------------------------------------------------
__global__ __launch_bounds__(256) void k_fc(const float* __restrict__ pool,
                                            const float* __restrict__ cnt,
                                            const float* __restrict__ Wfc,
                                            const float* __restrict__ bfc,
                                            float* __restrict__ out) {
    int g = threadIdx.x >> 2, o = threadIdx.x & 3;
    float inv = 1.0f / fmaxf(cnt[g], 1.0f);
    float s = 0.0f;
    #pragma unroll 8
    for (int k = 0; k < HID2; ++k)
        s += pool[(size_t)g * HID2 + k] * Wfc[k * OUT_DIM + o];
    out[g * OUT_DIM + o] = s * inv + bfc[o];
}

// ---------------------------------------------------------------------------
// Launch
// ---------------------------------------------------------------------------
static inline size_t alignup(size_t x) { return (x + 255) & ~(size_t)255; }

extern "C" void kernel_launch(void* const* d_in, const int* in_sizes, int n_in,
                              void* d_out, int out_size, void* d_ws, size_t ws_size,
                              hipStream_t stream) {
    const float* x     = (const float*)d_in[0];
    const int*   src   = (const int*)d_in[1];
    const int*   dst   = (const int*)d_in[2];
    const int*   batch = (const int*)d_in[3];
    const float* W1    = (const float*)d_in[4];
    const float* b1    = (const float*)d_in[5];
    const float* W2    = (const float*)d_in[6];
    const float* b2    = (const float*)d_in[7];
    const float* Wfc   = (const float*)d_in[8];
    const float* bfc   = (const float*)d_in[9];

    const int nNodes = in_sizes[0] / IN_DIM;   // 50000
    const int nEdges = in_sizes[1];            // 640000
    const int mtiles = nNodes / 16;            // 3125 (exact)

    // Workspace carve-up
    char* p = (char*)d_ws;
    _Float16* x16  = (_Float16*)p; p += alignup((size_t)nNodes * IN_DIM * 2);
    _Float16* w1t  = (_Float16*)p; p += alignup((size_t)IN_DIM * HID * 2);
    _Float16* w2t  = (_Float16*)p; p += alignup((size_t)HID * HID2 * 2);
    _Float16* h16  = (_Float16*)p; p += alignup((size_t)nNodes * HID * 2);
    float*    hlin = (float*)p;    p += alignup((size_t)nNodes * HID * 4);  // h1 f32, reused as h2 f32
    float*    agg  = (float*)p;    p += alignup((size_t)nNodes * HID * 4);  // reused for layer 2
    float*    deg  = (float*)p;    p += alignup((size_t)nNodes * 4);
    float*    dinv = (float*)p;    p += alignup((size_t)nNodes * 4);
    float*    pool = (float*)p;    p += (size_t)N_GRAPHS * HID2 * 4;
    float*    cnt  = (float*)p;    p += (size_t)N_GRAPHS * 4;

    const int B = 256;
    auto blocks = [](long long n, int b) { return (int)((n + b - 1) / b); };

    // --- degree / dinv ---
    hipMemsetAsync(deg, 0, (size_t)nNodes * 4, stream);
    k_degree<<<blocks(nEdges, B), B, 0, stream>>>(dst, deg, nEdges);
    k_dinv<<<blocks(nNodes, B), B, 0, stream>>>(deg, dinv, nNodes);

    // --- conversions ---
    k_cvt_f16<<<blocks((long long)nNodes * IN_DIM, B), B, 0, stream>>>(x, x16, nNodes * IN_DIM);
    k_cvt_wt<IN_DIM, HID><<<blocks(IN_DIM * HID, B), B, 0, stream>>>(W1, w1t);
    k_cvt_wt<HID, HID2><<<blocks(HID * HID2, B), B, 0, stream>>>(W2, w2t);

    // --- layer 1: GEMM -> scatter -> finalize(f16) ---
    k_wmma_gemm<IN_DIM, HID><<<blocks(mtiles, 8), B, 0, stream>>>(x16, w1t, hlin, mtiles);
    hipMemsetAsync(agg, 0, (size_t)nNodes * HID * 4, stream);
    k_scatter<HID><<<blocks((long long)nEdges * (HID / 4), B), B, 0, stream>>>(
        hlin, src, dst, dinv, agg, nEdges);
    k_finalize_f16<HID><<<blocks((long long)nNodes * HID, B), B, 0, stream>>>(
        agg, hlin, dinv, b1, h16, nNodes);

    // --- layer 2: GEMM -> scatter -> finalize + pool ---
    k_wmma_gemm<HID, HID2><<<blocks(mtiles, 8), B, 0, stream>>>(h16, w2t, hlin, mtiles);
    hipMemsetAsync(agg, 0, (size_t)nNodes * HID2 * 4, stream);
    k_scatter<HID2><<<blocks((long long)nEdges * (HID2 / 4), B), B, 0, stream>>>(
        hlin, src, dst, dinv, agg, nEdges);

    hipMemsetAsync(pool, 0, ((size_t)N_GRAPHS * HID2 + N_GRAPHS) * 4, stream); // pool + cnt
    k_count<<<blocks(nNodes, B), B, 0, stream>>>(batch, cnt, nNodes);
    k_finalize_pool<<<blocks((long long)nNodes * HID2, B), B, 0, stream>>>(
        agg, hlin, dinv, b2, batch, pool, nNodes);

    // --- final FC ---
    k_fc<<<1, 256, 0, stream>>>(pool, cnt, Wfc, bfc, (float*)d_out);
}
